// Head_61546881352364
// MI455X (gfx1250) — matrix operand order repaired and checked by reading, not compiled, not run on hardware
//
#include <hip/hip_runtime.h>
#include <hip/hip_bf16.h>

typedef _Float16 half_t;
typedef __attribute__((ext_vector_type(16))) _Float16 v16h;
typedef __attribute__((ext_vector_type(8)))  _Float16 v8h;
typedef __attribute__((ext_vector_type(8)))  float    v8f;
typedef __attribute__((ext_vector_type(4)))  float    v4f;

// ---------------------------------------------------------------------------
// Kernel 0: transpose + fp32->fp16 convert the weight matrices.
// W [C,H] fp32 row-major  ->  Wt [H,C] fp16 row-major
// ---------------------------------------------------------------------------
__global__ void weight_tr_kernel(const float* __restrict__ W,
                                 half_t* __restrict__ Wt,
                                 int Cdim, int Hdim) {
    int idx = blockIdx.x * blockDim.x + threadIdx.x;   // over H*C, k fastest
    if (idx >= Cdim * Hdim) return;
    int hcol = idx / Cdim;
    int k    = idx % Cdim;
    Wt[idx] = (half_t)W[(size_t)k * Hdim + hcol];
}

// ---------------------------------------------------------------------------
// Kernel 1: projection GEMM  X[M,C] (fp32) @ Wt[H,C]^T (fp16) -> Out (fp16)
// One wave computes one 16x16 output tile with v_wmma_f32_16x16x32_f16.
// transpose_out = 0 : Out is [M,H] row-major (Q, K)
// transpose_out = 1 : Out is [B,H,T]         (V, for contiguous P*V B-frags)
// ---------------------------------------------------------------------------
__global__ __launch_bounds__(256) void proj_kernel(
    const float* __restrict__ X, const half_t* __restrict__ Wt,
    half_t* __restrict__ Out, int M, int Cdim, int Hdim,
    int transpose_out, int Tdim)
{
    const int lane     = threadIdx.x & 31;
    const int wave     = threadIdx.x >> 5;
    const int half_sel = lane >> 4;        // 0: lanes 0-15, 1: lanes 16-31
    const int lane16   = lane & 15;

    const int ntiles_n = Hdim >> 4;                    // 8
    const int tile     = blockIdx.x * 8 + wave;
    const int mtile    = tile / ntiles_n;
    const int ntile    = tile % ntiles_n;
    if (mtile * 16 >= M) return;

    const float*  Arow = X  + (size_t)(mtile * 16 + lane16) * Cdim;
    const half_t* Brow = Wt + (size_t)(ntile * 16 + lane16) * Cdim;

    v8f acc = {};
    for (int kc = 0; kc < Cdim; kc += 32) {
        // A fragment (16x32 f16): lane row = lane16,
        // K = half_sel*16 + j (contiguous, matches B)
        const float* ap0 = Arow + kc + half_sel * 16;
        const float* ap1 = Arow + kc + half_sel * 16 + 8;
        v4f a0 = *(const v4f*)(ap0);
        v4f a1 = *(const v4f*)(ap0 + 4);
        v4f a2 = *(const v4f*)(ap1);
        v4f a3 = *(const v4f*)(ap1 + 4);
        v16h a;
#pragma unroll
        for (int j = 0; j < 4; ++j) {
            a[j]      = (half_t)a0[j];
            a[4 + j]  = (half_t)a1[j];
            a[8 + j]  = (half_t)a2[j];
            a[12 + j] = (half_t)a3[j];
        }
        // B fragment (32x16 f16): lane col = lane16,
        // K = half_sel*16 + 0..15  -> one contiguous 32B load
        v16h b = *(const v16h*)(Brow + kc + half_sel * 16);

        acc = __builtin_amdgcn_wmma_f32_16x16x32_f16(
            false, a, false, b, (short)0, acc, false, false);
    }

    // C/D layout: VGPR r, row = r + 8*half_sel, col = lane16
    if (!transpose_out) {
#pragma unroll
        for (int r = 0; r < 8; ++r) {
            int grow = mtile * 16 + r + 8 * half_sel;
            Out[(size_t)grow * Hdim + ntile * 16 + lane16] = (half_t)acc[r];
        }
    } else {
#pragma unroll
        for (int r = 0; r < 8; ++r) {
            int grow = mtile * 16 + r + 8 * half_sel;
            int bb = grow / Tdim, tt = grow % Tdim;
            Out[((size_t)bb * Hdim + ntile * 16 + lane16) * Tdim + tt] =
                (half_t)acc[r];
        }
    }
}

// ---------------------------------------------------------------------------
// Kernel 2: flash attention. One wave per (batch, 16-row query tile).
// Qh,Kh: [B*T,H] f16 ; Vt: [B,H,T] f16 ; Out: [B,T,H] f32
// ---------------------------------------------------------------------------
__global__ __launch_bounds__(256) void attn_kernel(
    const half_t* __restrict__ Qh, const half_t* __restrict__ Kh,
    const half_t* __restrict__ Vt, float* __restrict__ Out,
    int Bn, int Tn, int Hn)
{
    __shared__ half_t plds[8 * 16 * 32];        // 1KB per wave: P relayout
    const int lane     = threadIdx.x & 31;
    const int wave     = threadIdx.x >> 5;
    const int half_sel = lane >> 4;
    const int lane16   = lane & 15;

    const int ttiles = Tn >> 4;                 // 128
    const int gw = blockIdx.x * 8 + wave;
    const int b  = gw / ttiles;
    const int i  = gw % ttiles;                 // query tile index
    if (b >= Bn) return;

    half_t* P = plds + wave * 16 * 32;

    // Q fragments for h = 0..127 (four 16x32 A-frags), contiguous loads
    const half_t* Qrow = Qh + (size_t)(b * Tn + i * 16 + lane16) * Hn;
    v16h qa[4];
#pragma unroll
    for (int c = 0; c < 4; ++c) {
        v8h q0 = *(const v8h*)(Qrow + c * 32 + half_sel * 16);
        v8h q1 = *(const v8h*)(Qrow + c * 32 + half_sel * 16 + 8);
#pragma unroll
        for (int j = 0; j < 8; ++j) { qa[c][j] = q0[j]; qa[c][8 + j] = q1[j]; }
    }

    v8f acc[8];
#pragma unroll
    for (int nt = 0; nt < 8; ++nt) acc[nt] = (v8f){};
    float mrun[8], lrun[8];
#pragma unroll
    for (int r = 0; r < 8; ++r) { mrun[r] = -3.0e38f; lrun[r] = 0.0f; }

    const float scale = 0.03125f;               // C^-0.5 = 1/sqrt(1024)
    const half_t* Kbase = Kh + (size_t)(b * Tn) * Hn;
    const half_t* Vb    = Vt + (size_t)b * Hn * Tn;
    const int nblocks = (i * 16 + 16 + 31) >> 5;   // 32 keys per block

    for (int jb = 0; jb < nblocks; ++jb) {
        const int c0 = jb * 32;

        // ---- S = Q * K^T for a 16x32 key block (two 16x16 tiles) ----
        v8f s[2] = {(v8f){}, (v8f){}};
#pragma unroll
        for (int t = 0; t < 2; ++t) {
            const half_t* Kr = Kbase + (size_t)(c0 + t * 16 + lane16) * Hn;
#pragma unroll
            for (int c = 0; c < 4; ++c) {
                v16h kb = *(const v16h*)(Kr + c * 32 + half_sel * 16);
                s[t] = __builtin_amdgcn_wmma_f32_16x16x32_f16(
                    false, qa[c], false, kb, (short)0, s[t], false, false);
            }
        }

        // ---- scale + causal mask ----
#pragma unroll
        for (int t = 0; t < 2; ++t) {
            int col = c0 + t * 16 + lane16;
#pragma unroll
            for (int r = 0; r < 8; ++r) {
                int row = i * 16 + r + 8 * half_sel;
                float sv = s[t][r] * scale;
                s[t][r] = (col > row) ? -1.0e30f : sv;
            }
        }

        // ---- online softmax update (row = r + 8*half_sel, 16-lane groups) --
#pragma unroll
        for (int r = 0; r < 8; ++r) {
            float m = fmaxf(s[0][r], s[1][r]);
            m = fmaxf(m, __shfl_xor(m, 1, 32));
            m = fmaxf(m, __shfl_xor(m, 2, 32));
            m = fmaxf(m, __shfl_xor(m, 4, 32));
            m = fmaxf(m, __shfl_xor(m, 8, 32));
            float mn = fmaxf(mrun[r], m);
            float al = __expf(mrun[r] - mn);
            float p0 = __expf(s[0][r] - mn);
            float p1 = __expf(s[1][r] - mn);
            s[0][r] = p0; s[1][r] = p1;
            float rs = p0 + p1;
            rs += __shfl_xor(rs, 1, 32);
            rs += __shfl_xor(rs, 2, 32);
            rs += __shfl_xor(rs, 4, 32);
            rs += __shfl_xor(rs, 8, 32);
            lrun[r] = lrun[r] * al + rs;
            mrun[r] = mn;
#pragma unroll
            for (int nt = 0; nt < 8; ++nt) acc[nt][r] *= al;
        }

        // ---- P: C-layout -> A-layout via LDS (wave-private, no barrier) ----
#pragma unroll
        for (int t = 0; t < 2; ++t)
#pragma unroll
            for (int r = 0; r < 8; ++r)
                P[(r + 8 * half_sel) * 32 + t * 16 + lane16] = (half_t)s[t][r];

        const half_t* pr = P + lane16 * 32 + half_sel * 16;
        v8h p0 = *(const v8h*)pr;
        v8h p1 = *(const v8h*)(pr + 8);
        v16h pa;
#pragma unroll
        for (int j = 0; j < 8; ++j) { pa[j] = p0[j]; pa[8 + j] = p1[j]; }

        // ---- O += P * V  (V transposed: contiguous B-frag loads) ----
#pragma unroll
        for (int nt = 0; nt < 8; ++nt) {
            const half_t* Vr =
                Vb + (size_t)(nt * 16 + lane16) * Tn + c0 + half_sel * 16;
            v16h vb = *(const v16h*)Vr;
            acc[nt] = __builtin_amdgcn_wmma_f32_16x16x32_f16(
                false, pa, false, vb, (short)0, acc[nt], false, false);
        }
    }

    // ---- epilogue: divide by softmax denominator, store f32 [B,T,H] ----
    const size_t obase = ((size_t)b * Tn + i * 16) * Hn;
#pragma unroll
    for (int nt = 0; nt < 8; ++nt)
#pragma unroll
        for (int r = 0; r < 8; ++r) {
            int row = r + 8 * half_sel;
            Out[obase + (size_t)row * Hn + nt * 16 + lane16] =
                acc[nt][r] / lrun[r];
        }
}

// ---------------------------------------------------------------------------
extern "C" void kernel_launch(void* const* d_in, const int* in_sizes, int n_in,
                              void* d_out, int out_size, void* d_ws,
                              size_t ws_size, hipStream_t stream) {
    constexpr int Bn = 8, Tn = 2048, Cn = 1024, Hn = 128;
    const float* k_in = (const float*)d_in[0];
    const float* q_in = (const float*)d_in[1];
    const float* v_in = (const float*)d_in[2];
    const float* Wk   = (const float*)d_in[3];
    const float* Wq   = (const float*)d_in[4];
    const float* Wv   = (const float*)d_in[5];
    float* out = (float*)d_out;

    // workspace layout (all 256B-aligned)
    char* ws = (char*)d_ws;
    size_t off = 0;
    half_t* Wtk = (half_t*)(ws + off); off += (size_t)Hn * Cn * 2;
    half_t* Wtq = (half_t*)(ws + off); off += (size_t)Hn * Cn * 2;
    half_t* Wtv = (half_t*)(ws + off); off += (size_t)Hn * Cn * 2;
    half_t* Qh  = (half_t*)(ws + off); off += (size_t)Bn * Tn * Hn * 2;
    half_t* Kh  = (half_t*)(ws + off); off += (size_t)Bn * Tn * Hn * 2;
    half_t* Vt  = (half_t*)(ws + off); off += (size_t)Bn * Tn * Hn * 2;

    // 0) weight transpose + f16 convert
    {
        int n = Cn * Hn;
        dim3 g((n + 255) / 256), blk(256);
        weight_tr_kernel<<<g, blk, 0, stream>>>(Wk, Wtk, Cn, Hn);
        weight_tr_kernel<<<g, blk, 0, stream>>>(Wq, Wtq, Cn, Hn);
        weight_tr_kernel<<<g, blk, 0, stream>>>(Wv, Wtv, Cn, Hn);
    }

    // 1) projections: M = B*T rows, 16x16 tile per wave, 8 waves per block
    {
        int M = Bn * Tn;
        int tiles = (M / 16) * (Hn / 16);          // 8192
        dim3 g(tiles / 8), blk(256);
        proj_kernel<<<g, blk, 0, stream>>>(q_in, Wtq, Qh, M, Cn, Hn, 0, Tn);
        proj_kernel<<<g, blk, 0, stream>>>(k_in, Wtk, Kh, M, Cn, Hn, 0, Tn);
        proj_kernel<<<g, blk, 0, stream>>>(v_in, Wtv, Vt, M, Cn, Hn, 1, Tn);
    }

    // 2) flash attention: one wave per (batch, query tile)
    {
        int waves = Bn * (Tn / 16);                // 1024
        dim3 g(waves / 8), blk(256);
        attn_kernel<<<g, blk, 0, stream>>>(Qh, Kh, Vt, out, Bn, Tn, Hn);
    }
}
